// PcDecoding_63462436766097
// MI455X (gfx1250) — compile-verified, matrix operand order checked by reading, not run on hardware
//
#include <hip/hip_runtime.h>

typedef _Float16 h16;
typedef __attribute__((ext_vector_type(16))) _Float16 v16h;
typedef __attribute__((ext_vector_type(8)))  _Float16 v8h;
typedef __attribute__((ext_vector_type(8)))  float    v8f;

#define BN_EPS 1e-5f

// --------------------------------------------------------------------------
// 3-nearest-neighbor search + inverse-distance weights.
// One thread per query point; search set staged through LDS in 256-pt chunks.
// NA is a multiple of 256 (256..16384), so every block is one batch.
// --------------------------------------------------------------------------
__global__ __launch_bounds__(256)
void knn3_kernel(const float* __restrict__ xyzA, const float* __restrict__ xyzB,
                 int NA, int SB, int* __restrict__ knnIdx, float* __restrict__ knnW)
{
    __shared__ float sx[256], sy[256], sz[256];
    const int t = blockIdx.x * 256 + threadIdx.x;       // 0 .. B*NA-1
    const int b = t / NA;
    const float ax = xyzA[3*t+0], ay = xyzA[3*t+1], az = xyzA[3*t+2];
    const float* xb = xyzB + (size_t)b * SB * 3;
    float d0 = 3.0e38f, d1 = 3.0e38f, d2 = 3.0e38f;
    int   i0 = 0, i1 = 0, i2 = 0;
    for (int s0 = 0; s0 < SB; s0 += 256) {
        const int chunk = (SB - s0 < 256) ? (SB - s0) : 256;
        __syncthreads();
        if ((int)threadIdx.x < chunk) {
            const float* p = xb + 3 * (s0 + threadIdx.x);
            sx[threadIdx.x] = p[0];
            sy[threadIdx.x] = p[1];
            sz[threadIdx.x] = p[2];
        }
        __syncthreads();
        for (int j = 0; j < chunk; ++j) {
            const float dx = ax - sx[j], dy = ay - sy[j], dz = az - sz[j];
            const float d  = dx*dx + dy*dy + dz*dz;
            const int   si = s0 + j;
            if (d < d2) {
                if (d < d1) {
                    d2 = d1; i2 = i1;
                    if (d < d0) { d1 = d0; i1 = i0; d0 = d; i0 = si; }
                    else        { d1 = d;  i1 = si; }
                } else { d2 = d; i2 = si; }
            }
        }
    }
    float w0 = 1.f/(d0 + 1e-8f), w1 = 1.f/(d1 + 1e-8f), w2 = 1.f/(d2 + 1e-8f);
    const float inv = 1.f/(w0 + w1 + w2);
    knnIdx[3*t+0] = i0; knnIdx[3*t+1] = i1; knnIdx[3*t+2] = i2;
    knnW [3*t+0] = w0*inv; knnW[3*t+1] = w1*inv; knnW[3*t+2] = w2*inv;
}

// --------------------------------------------------------------------------
// Weighted 3-point feature gather + skip-concat, output packed f16 activations
// X[row, 0:Cskip] = skip ;  X[row, Cskip:Cskip+Cf] = sum_j w_j * src[idx_j]
// --------------------------------------------------------------------------
template <typename SrcT>
__global__ __launch_bounds__(256)
void interp_concat_kernel(const float* __restrict__ skip, int Cskip,
                          const SrcT* __restrict__ src, int Cf, int SB,
                          const int* __restrict__ knnIdx, const float* __restrict__ knnW,
                          h16* __restrict__ Xout, int NA, int total)
{
    const int t = blockIdx.x * 256 + threadIdx.x;
    if (t >= total) return;
    const int Cin = Cskip + Cf;
    const int row = t / Cin;
    const int c   = t - row * Cin;
    float val;
    if (c < Cskip) {
        val = skip[(size_t)row * Cskip + c];
    } else {
        const int cf = c - Cskip;
        const int b  = row / NA;
        const int j0 = knnIdx[3*row+0], j1 = knnIdx[3*row+1], j2 = knnIdx[3*row+2];
        const float w0 = knnW[3*row+0], w1 = knnW[3*row+1], w2 = knnW[3*row+2];
        const SrcT* sb = src + (size_t)b * SB * Cf;
        val = w0 * (float)sb[(size_t)j0 * Cf + cf]
            + w1 * (float)sb[(size_t)j1 * Cf + cf]
            + w2 * (float)sb[(size_t)j2 * Cf + cf];
    }
    Xout[t] = (h16)val;
}

// --------------------------------------------------------------------------
// Fold BN into per-output-channel scale/bias and convert weights to f16.
//   scale = gamma * rsqrt(var + eps) ; bias = (b - mean)*scale + beta
// ga==nullptr -> plain linear (scale=1, bias=b).
// --------------------------------------------------------------------------
__global__ __launch_bounds__(256)
void fold_params_kernel(const float* __restrict__ W, const float* __restrict__ b,
                        const float* __restrict__ ga, const float* __restrict__ be,
                        const float* __restrict__ m,  const float* __restrict__ v,
                        h16* __restrict__ Wh, float* __restrict__ sc, float* __restrict__ bs,
                        int Cout, int Cin)
{
    const int t = blockIdx.x * 256 + threadIdx.x;
    if (t < Cout * Cin) Wh[t] = (h16)W[t];
    if (t < Cout) {
        if (ga) {
            const float s = ga[t] * rsqrtf(v[t] + BN_EPS);
            sc[t] = s;
            bs[t] = (b[t] - m[t]) * s + be[t];
        } else {
            sc[t] = 1.f;
            bs[t] = b[t];
        }
    }
}

// --------------------------------------------------------------------------
// WMMA GEMM:  Y[r,o] = post( sum_k X[r,k]*W[o,k] )  with per-column scale/bias
// and optional ReLU.  f16 inputs, f32 accumulate via v_wmma_f32_16x16x32_f16.
// One wave computes a 16x64 tile (4 accumulators, shared A fragment).
// Per ISA 7.12.2: lane L -> A row = L&15, K-halves {kb..kb+7, 16+kb..16+kb+7},
// kb = (L>=16)*8  ->  fragment = two contiguous 16B loads.  B symmetric on the
// transposed weight row.  C/D: VGPR r -> M = r + 8*(L>=16), N = L&15.
// Requires: R %16==0, Cin %32==0, Cout %64==0 (all shapes here comply).
// --------------------------------------------------------------------------
template <typename OutT>
__global__ __launch_bounds__(256)
void gemm_wmma_kernel(const h16* __restrict__ X, const h16* __restrict__ Wh,
                      const float* __restrict__ sc, const float* __restrict__ bs,
                      OutT* __restrict__ Y, int R, int Cin, int Cout, int relu)
{
    const int lane  = threadIdx.x & 31;
    const int wid   = threadIdx.x >> 5;
    const int npass = Cout >> 6;                 // 64-wide column groups
    const int total = (R >> 4) * npass;
    const int gw    = blockIdx.x * 8 + wid;      // wave-uniform -> EXEC all-1s
    if (gw >= total) return;
    const int mt   = gw / npass;
    const int np   = gw - mt * npass;
    const int row0 = mt << 4;
    const int n0   = np << 6;

    const int m  = lane & 15;                    // A row / B,C column in tile
    const int kb = (lane >> 4) << 3;             // K sub-offset 0 or 8

    v8f acc[4] = {};
    const h16* xrow = X + (size_t)(row0 + m) * Cin + kb;

    for (int k0 = 0; k0 < Cin; k0 += 32) {
        v16h a;
        {
            v8h lo = *(const v8h*)(xrow + k0);
            v8h hi = *(const v8h*)(xrow + k0 + 16);
#pragma unroll
            for (int i = 0; i < 8; ++i) { a[i] = lo[i]; a[i+8] = hi[i]; }
        }
#pragma unroll
        for (int nt = 0; nt < 4; ++nt) {
            const h16* wrow = Wh + (size_t)(n0 + nt*16 + m) * Cin + kb + k0;
            v16h bf;
            {
                v8h lo = *(const v8h*)(wrow);
                v8h hi = *(const v8h*)(wrow + 16);
#pragma unroll
                for (int i = 0; i < 8; ++i) { bf[i] = lo[i]; bf[i+8] = hi[i]; }
            }
            acc[nt] = __builtin_amdgcn_wmma_f32_16x16x32_f16(
                false, a, false, bf, (short)0, acc[nt], false, false);
        }
    }

    const int mbase = (lane >> 4) << 3;          // 0 or 8
#pragma unroll
    for (int nt = 0; nt < 4; ++nt) {
        const int col = n0 + nt*16 + m;
        const float s = sc[col], z = bs[col];
#pragma unroll
        for (int r = 0; r < 8; ++r) {
            float y = acc[nt][r] * s + z;
            if (relu) y = fmaxf(y, 0.f);
            Y[(size_t)(row0 + mbase + r) * Cout + col] = (OutT)y;
        }
    }
}

// --------------------------------------------------------------------------
// Host orchestration
// --------------------------------------------------------------------------
extern "C" void kernel_launch(void* const* d_in, const int* in_sizes, int n_in,
                              void* d_out, int out_size, void* d_ws, size_t ws_size,
                              hipStream_t stream)
{
    (void)in_sizes; (void)n_in; (void)out_size; (void)ws_size;
    const int Bb = 4;

    const float* xyz0    = (const float*)d_in[0];   // (4,16384,3)
    const float* xyz1    = (const float*)d_in[1];   // (4, 4096,3)
    const float* xyz2    = (const float*)d_in[2];   // (4, 1024,3)
    const float* xyz3    = (const float*)d_in[3];   // (4,  256,3)
    const float* xyz4    = (const float*)d_in[4];   // (4,   64,3)
    const float* points1 = (const float*)d_in[5];   // (4, 4096, 32)
    const float* points2 = (const float*)d_in[6];   // (4, 1024, 64)
    const float* points3 = (const float*)d_in[7];   // (4,  256,128)
    const float* points4 = (const float*)d_in[8];   // (4,   64,256)

    // params pytree leaves (sorted dict keys): conv2_W, conv2_b, fp1, fp2, fp3, fp4, head1
    const float* conv2_W = (const float*)d_in[9];
    const float* conv2_b = (const float*)d_in[10];
    const int fp1_base  = 11;                       // 3 layers * 6 leaves
    const int fp2_base  = 29;                       // 2 layers
    const int fp3_base  = 41;
    const int fp4_base  = 53;
    const int head_base = 65;
    auto P = [&](int base, int layer, int which) -> const float* {
        return (const float*)d_in[base + layer*6 + which];
    };

    // ---- workspace carve (bump allocator, 256B aligned) ----
    char* wsp = (char*)d_ws;
    auto walloc = [&](size_t bytes) -> void* {
        void* p = (void*)wsp;
        wsp += (bytes + 255) & ~(size_t)255;
        return p;
    };
    int*   knnIdx = (int*)  walloc((size_t)65536 * 3 * sizeof(int));
    float* knnW   = (float*)walloc((size_t)65536 * 3 * sizeof(float));
    h16*   ping   = (h16*)  walloc((size_t)65536 * 128 * sizeof(h16));   // max rows*chan
    h16*   pong   = (h16*)  walloc((size_t)65536 * 128 * sizeof(h16));
    h16*   l3buf  = (h16*)  walloc((size_t)1024  * 256 * sizeof(h16));
    h16*   l2buf  = (h16*)  walloc((size_t)4096  * 256 * sizeof(h16));
    h16*   l1buf  = (h16*)  walloc((size_t)16384 * 128 * sizeof(h16));
    h16*   Wh     = (h16*)  walloc((size_t)256 * 384 * sizeof(h16));
    float* scbuf  = (float*)walloc(256 * sizeof(float));
    float* bsbuf  = (float*)walloc(256 * sizeof(float));

    // ---- helpers ----
    auto run_interp = [&](const float* xa, const float* xb, int NA, int SB,
                          const float* skip, int Cskip,
                          const void* src, bool srcHalf, int Cf, h16* Xout) {
        knn3_kernel<<<(Bb * NA) / 256, 256, 0, stream>>>(xa, xb, NA, SB, knnIdx, knnW);
        const int total = Bb * NA * (Cskip + Cf);
        const int blocks = (total + 255) / 256;
        if (srcHalf)
            interp_concat_kernel<h16><<<blocks, 256, 0, stream>>>(
                skip, Cskip, (const h16*)src, Cf, SB, knnIdx, knnW, Xout, NA, total);
        else
            interp_concat_kernel<float><<<blocks, 256, 0, stream>>>(
                skip, Cskip, (const float*)src, Cf, SB, knnIdx, knnW, Xout, NA, total);
    };

    auto run_layer = [&](const h16* Xin, int R, int Cin, int Cout,
                         const float* W, const float* b, const float* ga,
                         const float* be, const float* m, const float* v,
                         void* Yout, bool outHalf, bool relu) {
        fold_params_kernel<<<(Cout * Cin + 255) / 256, 256, 0, stream>>>(
            W, b, ga, be, m, v, Wh, scbuf, bsbuf, Cout, Cin);
        const int waves  = (R / 16) * (Cout / 64);
        const int blocks = (waves + 7) / 8;
        if (outHalf)
            gemm_wmma_kernel<h16><<<blocks, 256, 0, stream>>>(
                Xin, Wh, scbuf, bsbuf, (h16*)Yout, R, Cin, Cout, relu ? 1 : 0);
        else
            gemm_wmma_kernel<float><<<blocks, 256, 0, stream>>>(
                Xin, Wh, scbuf, bsbuf, (float*)Yout, R, Cin, Cout, relu ? 1 : 0);
    };
    auto run_bn_layer = [&](const h16* Xin, int R, int Cin, int Cout,
                            int base, int layer, void* Yout, bool outHalf) {
        run_layer(Xin, R, Cin, Cout,
                  P(base, layer, 0), P(base, layer, 1), P(base, layer, 2),
                  P(base, layer, 3), P(base, layer, 4), P(base, layer, 5),
                  Yout, outHalf, true);
    };

    // ---- fp4: xyz3(256) <- xyz4(64); concat(points3[128], interp(points4[256])) -> MLP[384->256->256]
    run_interp(xyz3, xyz4, 256, 64, points3, 128, points4, false, 256, ping);
    run_bn_layer(ping, Bb * 256, 384, 256, fp4_base, 0, pong, true);
    run_bn_layer(pong, Bb * 256, 256, 256, fp4_base, 1, l3buf, true);

    // ---- fp3: xyz2(1024) <- xyz3(256); concat(points2[64], interp(l3[256])) -> MLP[320->256->256]
    run_interp(xyz2, xyz3, 1024, 256, points2, 64, l3buf, true, 256, ping);
    run_bn_layer(ping, Bb * 1024, 320, 256, fp3_base, 0, pong, true);
    run_bn_layer(pong, Bb * 1024, 256, 256, fp3_base, 1, l2buf, true);

    // ---- fp2: xyz1(4096) <- xyz2(1024); concat(points1[32], interp(l2[256])) -> MLP[288->256->128]
    run_interp(xyz1, xyz2, 4096, 1024, points1, 32, l2buf, true, 256, ping);
    run_bn_layer(ping, Bb * 4096, 288, 256, fp2_base, 0, pong, true);
    run_bn_layer(pong, Bb * 4096, 256, 128, fp2_base, 1, l1buf, true);

    // ---- fp1: xyz0(16384) <- xyz1(4096); interp(l1[128]) -> MLP[128->128->128->128]
    run_interp(xyz0, xyz1, 16384, 4096, nullptr, 0, l1buf, true, 128, ping);
    run_bn_layer(ping, Bb * 16384, 128, 128, fp1_base, 0, pong, true);
    run_bn_layer(pong, Bb * 16384, 128, 128, fp1_base, 1, ping, true);
    run_bn_layer(ping, Bb * 16384, 128, 128, fp1_base, 2, pong, true);

    // ---- head1 (128->128, BN+ReLU) then conv2 (128->128, bias only) -> f32 output
    run_bn_layer(pong, Bb * 16384, 128, 128, head_base, 0, ping, true);
    run_layer(ping, Bb * 16384, 128, 128, conv2_W, conv2_b,
              nullptr, nullptr, nullptr, nullptr, d_out, false, false);
}
